// CausalSelfAttention_75316546502942
// MI455X (gfx1250) — compile-verified, matrix operand order checked by reading, not run on hardware
//
#include <hip/hip_runtime.h>

// ---------------------------------------------------------------------------
// CDNA5 (gfx1250) causal self-attention: qkv GEMM + flash attention + proj.
// Matrix math: v_wmma_f32_16x16x32_bf16 (wave32, f32 accumulate).
// Tile staging: global_load_async_to_lds_b128 (ASYNCcnt DMA path),
// double-buffered in the GEMMs.
// ---------------------------------------------------------------------------

typedef __bf16 bf16_t;
typedef __attribute__((ext_vector_type(16))) __bf16 v16bf;
typedef __attribute__((ext_vector_type(8)))  __bf16 v8bf;
typedef __attribute__((ext_vector_type(4)))  __bf16 v4bf;
typedef __attribute__((ext_vector_type(8)))  float  v8f;

struct Frag { union { v16bf v; v8bf h[2]; }; };

__device__ __forceinline__ v8f vzero8() {
    v8f z = {0.f, 0.f, 0.f, 0.f, 0.f, 0.f, 0.f, 0.f};
    return z;
}

// D = A(16x32) * B(32x16) + C, bf16 inputs, f32 accumulate
__device__ __forceinline__ v8f wmma_bf16(v16bf a, v16bf b, v8f c) {
    return __builtin_amdgcn_wmma_f32_16x16x32_bf16(
        false, a, false, b, (short)0, c, false, false);
}

// Async DMA: 16 bytes global -> LDS per lane (ASYNCcnt-tracked, no VGPR data).
// LDS aperture maps flat addr[31:0] directly to the LDS offset (ISA 10.2).
__device__ __forceinline__ void async_ld16(bf16_t* lds, const bf16_t* g) {
    unsigned ldsOff = (unsigned)(size_t)lds;
    unsigned long long ga = (unsigned long long)(size_t)g;
    asm volatile("global_load_async_to_lds_b128 %0, %1, off"
                 :: "v"(ldsOff), "v"(ga) : "memory");
}
__device__ __forceinline__ void async_wait0() {
    asm volatile("s_wait_asynccnt 0" ::: "memory");
}

// ---------------------------------------------------------------------------
// Kernel 0: fp32 -> bf16 conversion (vectorized x4)
// ---------------------------------------------------------------------------
__global__ __launch_bounds__(256) void cvt_kernel(const float4* __restrict__ src,
                                                  v4bf* __restrict__ dst, int n4) {
    int i = blockIdx.x * 256 + threadIdx.x;
    if (i < n4) {
        float4 f = src[i];
        v4bf o;
        o[0] = (bf16_t)f.x; o[1] = (bf16_t)f.y;
        o[2] = (bf16_t)f.z; o[3] = (bf16_t)f.w;
        dst[i] = o;
    }
}

// ---------------------------------------------------------------------------
// GEMM tiling: WG = 256 threads = 8 waves in 4(M) x 2(N) grid.
// WG tile 256(M) x 128(N), wave tile 64x64 (4x4 WMMA accs), K-step 32.
// Double-buffered LDS, async DMA staging.
// ---------------------------------------------------------------------------
#define LDT 40   // padded LDS row stride in bf16 elems (80B, 16B aligned)
#define TM  256
#define TN  128

// ---------------------------------------------------------------------------
// Kernel 1: qkv = x @ w_qkv^T  (8192 x 6144, K=2048), fused split epilogue.
//   q  -> bf16 [B,H,T,Dh]         (workspace, attention A operand)
//   k  -> f32  [B,H,T,Dh] out  +  bf16 [B,H,T,Dh]  (attention B operand)
//   v  -> f32  [B,H,T,Dh] out  +  bf16 [B,H,Dh,T]  (transposed, B operand)
// ---------------------------------------------------------------------------
__global__ __launch_bounds__(256) void qkv_gemm_kernel(
    const bf16_t* __restrict__ A,    // xb     [8192 x 2048]
    const bf16_t* __restrict__ Bm,   // wqkvb  [6144 x 2048]
    float* __restrict__ outK, float* __restrict__ outV,
    bf16_t* __restrict__ qb, bf16_t* __restrict__ kb, bf16_t* __restrict__ vtb)
{
    __shared__ bf16_t sA[2][TM * LDT];   // 2 x 20 KB
    __shared__ bf16_t sB[2][TN * LDT];   // 2 x 10 KB

    const int K = 2048;
    int tid = threadIdx.x;
    int lane = tid & 31, wave = tid >> 5;
    int l15 = lane & 15, half = lane >> 4;
    int waveM = wave >> 1, waveN = wave & 1;     // 4 x 2 wave grid
    int m0 = blockIdx.y * TM, n0 = blockIdx.x * TN;

    auto stage = [&](int buf, int kk) {
#pragma unroll
        for (int it = 0; it < 4; ++it) {         // A: 1024 chunks of 8 bf16
            int c = tid + it * 256;
            int row = c >> 2, cc = (c & 3) * 8;
            async_ld16(&sA[buf][row * LDT + cc],
                       &A[(size_t)(m0 + row) * K + kk + cc]);
        }
#pragma unroll
        for (int it = 0; it < 2; ++it) {         // B: 512 chunks
            int c = tid + it * 256;
            int row = c >> 2, cc = (c & 3) * 8;
            async_ld16(&sB[buf][row * LDT + cc],
                       &Bm[(size_t)(n0 + row) * K + kk + cc]);
        }
    };

    v8f acc[4][4];
#pragma unroll
    for (int mt = 0; mt < 4; ++mt)
#pragma unroll
        for (int nt = 0; nt < 4; ++nt) acc[mt][nt] = vzero8();

    stage(0, 0);
    async_wait0();
    __syncthreads();

    for (int kk = 0; kk < K; kk += 32) {
        int cur = (kk >> 5) & 1;
        if (kk + 32 < K) stage(cur ^ 1, kk + 32);   // prefetch next tile

        Frag af[4], bfr[4];
#pragma unroll
        for (int mt = 0; mt < 4; ++mt) {
            int ar = waveM * 64 + mt * 16 + l15;
            af[mt].h[0] = *(const v8bf*)&sA[cur][ar * LDT + half * 8];
            af[mt].h[1] = *(const v8bf*)&sA[cur][ar * LDT + 16 + half * 8];
        }
#pragma unroll
        for (int nt = 0; nt < 4; ++nt) {
            int br = waveN * 64 + nt * 16 + l15;
            bfr[nt].h[0] = *(const v8bf*)&sB[cur][br * LDT + half * 8];
            bfr[nt].h[1] = *(const v8bf*)&sB[cur][br * LDT + 16 + half * 8];
        }
#pragma unroll
        for (int mt = 0; mt < 4; ++mt)
#pragma unroll
            for (int nt = 0; nt < 4; ++nt)
                acc[mt][nt] = wmma_bf16(af[mt].v, bfr[nt].v, acc[mt][nt]);

        async_wait0();
        __syncthreads();
    }

    // split epilogue (section is uniform per 16-wide subtile)
#pragma unroll
    for (int mt = 0; mt < 4; ++mt) {
#pragma unroll
        for (int nt = 0; nt < 4; ++nt) {
            int col = n0 + waveN * 64 + nt * 16 + l15;
            int sec = col >> 11;                 // 0=q 1=k 2=v
            int jj = col & 2047;
            int h = jj >> 7, d = jj & 127;
#pragma unroll
            for (int r = 0; r < 8; ++r) {
                int mrow = m0 + waveM * 64 + mt * 16 + half * 8 + r;
                int b = mrow >> 11, t = mrow & 2047;
                float val = acc[mt][nt][r];
                int idx = ((b * 16 + h) * 2048 + t) * 128 + d;
                if (sec == 0) {
                    qb[idx] = (bf16_t)val;
                } else if (sec == 1) {
                    outK[idx] = val;
                    kb[idx] = (bf16_t)val;
                } else {
                    outV[idx] = val;
                    vtb[((b * 16 + h) * 128 + d) * 2048 + t] = (bf16_t)val;
                }
            }
        }
    }
}

// ---------------------------------------------------------------------------
// Kernel 2: flash attention per (b,h). 64 q-rows / WG, 4 waves x 16 rows.
// Online softmax, causal mask on diagonal block only. Async K/V staging.
// ---------------------------------------------------------------------------
#define LKS 136  // K tile LDS stride (64 rows x 128 d)
#define LVS 72   // V^T tile LDS stride (128 d-rows x 64 s) and P stride

__global__ __launch_bounds__(128) void attn_kernel(
    const bf16_t* __restrict__ qb, const bf16_t* __restrict__ kb,
    const bf16_t* __restrict__ vtb, bf16_t* __restrict__ yb)
{
    const int T = 2048, Dh = 128, H = 16;
    __shared__ bf16_t sK[64 * LKS];      // [s][d]
    __shared__ bf16_t sV[128 * LVS];     // [d][s]
    __shared__ bf16_t sP[4 * 16 * LVS];  // per-wave P staging [t][s]

    int tid = threadIdx.x;
    int lane = tid & 31, wave = tid >> 5;
    int l15 = lane & 15, half = lane >> 4;
    int qblk = blockIdx.x, h = blockIdx.y, b = blockIdx.z;
    int bh = b * H + h;
    const bf16_t* qB = qb + (size_t)bh * T * Dh;
    const bf16_t* kB = kb + (size_t)bh * T * Dh;
    const bf16_t* vB = vtb + (size_t)bh * Dh * T;

    int t0 = qblk * 64 + wave * 16;

    // Q fragments held in registers for the whole s-loop (16 x 128)
    Frag qf[4];
#pragma unroll
    for (int ks = 0; ks < 4; ++ks) {
        int row = t0 + l15;
        qf[ks].h[0] = *(const v8bf*)&qB[row * Dh + ks * 32 + half * 8];
        qf[ks].h[1] = *(const v8bf*)&qB[row * Dh + ks * 32 + 16 + half * 8];
    }

    v8f o[8];
    float mrow[8], lrow[8];
#pragma unroll
    for (int i = 0; i < 8; ++i) { o[i] = vzero8(); mrow[i] = -1e30f; lrow[i] = 0.f; }

    const float scale = 0.08838834764831845f;   // 1/sqrt(128)
    bf16_t* myP = &sP[wave * 16 * LVS];

    for (int sb = 0; sb <= qblk; ++sb) {
        int s0 = sb * 64;
        // async-stage K tile [64 x 128] and V^T tile [128 x 64]
        for (int c = tid; c < 1024; c += 128) {
            int kr = c >> 4, kc = (c & 15) * 8;
            async_ld16(&sK[kr * LKS + kc], &kB[(s0 + kr) * Dh + kc]);
            int vr = c >> 3, vc = (c & 7) * 8;
            async_ld16(&sV[vr * LVS + vc], &vB[vr * T + s0 + vc]);
        }
        async_wait0();
        __syncthreads();

        // S = Q K^T (16 x 64 per wave)
        v8f sacc[4];
#pragma unroll
        for (int nt = 0; nt < 4; ++nt) sacc[nt] = vzero8();
#pragma unroll
        for (int ks = 0; ks < 4; ++ks) {
#pragma unroll
            for (int nt = 0; nt < 4; ++nt) {
                Frag kf;
                int sr = nt * 16 + l15;
                kf.h[0] = *(const v8bf*)&sK[sr * LKS + ks * 32 + half * 8];
                kf.h[1] = *(const v8bf*)&sK[sr * LKS + ks * 32 + 16 + half * 8];
                sacc[nt] = wmma_bf16(qf[ks].v, kf.v, sacc[nt]);
            }
        }

        bool diag = (sb == qblk);
#pragma unroll
        for (int r = 0; r < 8; ++r) {
            int trow = t0 + half * 8 + r;
            float v0 = sacc[0][r] * scale;
            float v1 = sacc[1][r] * scale;
            float v2 = sacc[2][r] * scale;
            float v3 = sacc[3][r] * scale;
            if (diag) {
                if (s0 + 0  + l15 > trow) v0 = -1e30f;
                if (s0 + 16 + l15 > trow) v1 = -1e30f;
                if (s0 + 32 + l15 > trow) v2 = -1e30f;
                if (s0 + 48 + l15 > trow) v3 = -1e30f;
            }
            float mx = fmaxf(fmaxf(v0, v1), fmaxf(v2, v3));
            mx = fmaxf(mx, __shfl_xor(mx, 1));
            mx = fmaxf(mx, __shfl_xor(mx, 2));
            mx = fmaxf(mx, __shfl_xor(mx, 4));
            mx = fmaxf(mx, __shfl_xor(mx, 8));
            float mnew = fmaxf(mrow[r], mx);
            float al = __expf(mrow[r] - mnew);
            mrow[r] = mnew;
            float p0 = __expf(v0 - mnew), p1 = __expf(v1 - mnew);
            float p2 = __expf(v2 - mnew), p3 = __expf(v3 - mnew);
            float sum = p0 + p1 + p2 + p3;
            sum += __shfl_xor(sum, 1);
            sum += __shfl_xor(sum, 2);
            sum += __shfl_xor(sum, 4);
            sum += __shfl_xor(sum, 8);
            lrow[r] = lrow[r] * al + sum;
#pragma unroll
            for (int dt = 0; dt < 8; ++dt) o[dt][r] *= al;
            int pr = (half * 8 + r) * LVS + l15;
            myP[pr +  0] = (bf16_t)p0;
            myP[pr + 16] = (bf16_t)p1;
            myP[pr + 32] = (bf16_t)p2;
            myP[pr + 48] = (bf16_t)p3;
        }

        // O += P V  (16 x 128 per wave), P reloaded in A-fragment layout
#pragma unroll
        for (int ks2 = 0; ks2 < 2; ++ks2) {
            Frag pf;
            pf.h[0] = *(const v8bf*)&myP[l15 * LVS + ks2 * 32 + half * 8];
            pf.h[1] = *(const v8bf*)&myP[l15 * LVS + ks2 * 32 + 16 + half * 8];
#pragma unroll
            for (int dt = 0; dt < 8; ++dt) {
                Frag vf;
                int dr = dt * 16 + l15;
                vf.h[0] = *(const v8bf*)&sV[dr * LVS + ks2 * 32 + half * 8];
                vf.h[1] = *(const v8bf*)&sV[dr * LVS + ks2 * 32 + 16 + half * 8];
                o[dt] = wmma_bf16(pf.v, vf.v, o[dt]);
            }
        }
        __syncthreads();
    }

    // epilogue: normalize and store y (bf16, [B,T,C] layout for proj GEMM)
#pragma unroll
    for (int r = 0; r < 8; ++r) {
        float inv = 1.0f / lrow[r];
        int trow = t0 + half * 8 + r;
        size_t base = ((size_t)(b * T + trow)) * 2048 + h * Dh;
#pragma unroll
        for (int dt = 0; dt < 8; ++dt)
            yb[base + dt * 16 + l15] = (bf16_t)(o[dt][r] * inv);
    }
}

// ---------------------------------------------------------------------------
// Kernel 3: out_y = yb @ w_proj^T  (8192 x 2048, K=2048), fp32 output.
// ---------------------------------------------------------------------------
__global__ __launch_bounds__(256) void proj_gemm_kernel(
    const bf16_t* __restrict__ A,    // yb     [8192 x 2048]
    const bf16_t* __restrict__ Bm,   // wprojb [2048 x 2048]
    float* __restrict__ out)
{
    __shared__ bf16_t sA[2][TM * LDT];
    __shared__ bf16_t sB[2][TN * LDT];

    const int K = 2048;
    int tid = threadIdx.x;
    int lane = tid & 31, wave = tid >> 5;
    int l15 = lane & 15, half = lane >> 4;
    int waveM = wave >> 1, waveN = wave & 1;
    int m0 = blockIdx.y * TM, n0 = blockIdx.x * TN;

    auto stage = [&](int buf, int kk) {
#pragma unroll
        for (int it = 0; it < 4; ++it) {
            int c = tid + it * 256;
            int row = c >> 2, cc = (c & 3) * 8;
            async_ld16(&sA[buf][row * LDT + cc],
                       &A[(size_t)(m0 + row) * K + kk + cc]);
        }
#pragma unroll
        for (int it = 0; it < 2; ++it) {
            int c = tid + it * 256;
            int row = c >> 2, cc = (c & 3) * 8;
            async_ld16(&sB[buf][row * LDT + cc],
                       &Bm[(size_t)(n0 + row) * K + kk + cc]);
        }
    };

    v8f acc[4][4];
#pragma unroll
    for (int mt = 0; mt < 4; ++mt)
#pragma unroll
        for (int nt = 0; nt < 4; ++nt) acc[mt][nt] = vzero8();

    stage(0, 0);
    async_wait0();
    __syncthreads();

    for (int kk = 0; kk < K; kk += 32) {
        int cur = (kk >> 5) & 1;
        if (kk + 32 < K) stage(cur ^ 1, kk + 32);

        Frag af[4], bfr[4];
#pragma unroll
        for (int mt = 0; mt < 4; ++mt) {
            int ar = waveM * 64 + mt * 16 + l15;
            af[mt].h[0] = *(const v8bf*)&sA[cur][ar * LDT + half * 8];
            af[mt].h[1] = *(const v8bf*)&sA[cur][ar * LDT + 16 + half * 8];
        }
#pragma unroll
        for (int nt = 0; nt < 4; ++nt) {
            int br = waveN * 64 + nt * 16 + l15;
            bfr[nt].h[0] = *(const v8bf*)&sB[cur][br * LDT + half * 8];
            bfr[nt].h[1] = *(const v8bf*)&sB[cur][br * LDT + 16 + half * 8];
        }
#pragma unroll
        for (int mt = 0; mt < 4; ++mt)
#pragma unroll
            for (int nt = 0; nt < 4; ++nt)
                acc[mt][nt] = wmma_bf16(af[mt].v, bfr[nt].v, acc[mt][nt]);

        async_wait0();
        __syncthreads();
    }

#pragma unroll
    for (int mt = 0; mt < 4; ++mt) {
#pragma unroll
        for (int nt = 0; nt < 4; ++nt) {
            int col = n0 + waveN * 64 + nt * 16 + l15;
#pragma unroll
            for (int r = 0; r < 8; ++r) {
                int mrow = m0 + waveM * 64 + mt * 16 + half * 8 + r;
                out[(size_t)mrow * 2048 + col] = acc[mt][nt][r];
            }
        }
    }
}

// ---------------------------------------------------------------------------
// Host launcher
// ---------------------------------------------------------------------------
extern "C" void kernel_launch(void* const* d_in, const int* in_sizes, int n_in,
                              void* d_out, int out_size, void* d_ws, size_t ws_size,
                              hipStream_t stream) {
    const int B = 4, T = 2048, C = 2048, H = 16, Dh = 128;
    const size_t BTC = (size_t)B * T * C;          // 16,777,216

    const float* x     = (const float*)d_in[0];    // [B,T,C]
    const float* wqkv  = (const float*)d_in[1];    // [3C,C]
    const float* wproj = (const float*)d_in[2];    // [C,C]

    float* outY = (float*)d_out;                   // [B,T,C]
    float* outK = outY + BTC;                      // [B,H,T,Dh]
    float* outV = outK + BTC;                      // [B,H,T,Dh]

    // workspace carve-up (all bf16), ~192 MB total
    char* w = (char*)d_ws;
    bf16_t* xb     = (bf16_t*)w; w += BTC * 2;
    bf16_t* wqkvb  = (bf16_t*)w; w += (size_t)3 * C * C * 2;
    bf16_t* wprojb = (bf16_t*)w; w += (size_t)C * C * 2;
    bf16_t* qb     = (bf16_t*)w; w += BTC * 2;     // [B,H,T,Dh]
    bf16_t* kbw    = (bf16_t*)w; w += BTC * 2;     // [B,H,T,Dh]
    bf16_t* vtb    = (bf16_t*)w; w += BTC * 2;     // [B,H,Dh,T]
    bf16_t* yb     = (bf16_t*)w; w += BTC * 2;     // [B,T,C]
    (void)ws_size; (void)in_sizes; (void)n_in; (void)out_size;

    // 0) fp32 -> bf16
    {
        int n4 = (int)(BTC / 4);
        cvt_kernel<<<n4 / 256, 256, 0, stream>>>((const float4*)x, (v4bf*)xb, n4);
        int nq = 3 * C * C / 4;
        cvt_kernel<<<nq / 256, 256, 0, stream>>>((const float4*)wqkv, (v4bf*)wqkvb, nq);
        int np = C * C / 4;
        cvt_kernel<<<np / 256, 256, 0, stream>>>((const float4*)wproj, (v4bf*)wprojb, np);
    }

    // 1) qkv GEMM with fused split (q/k/v + transposed v)
    qkv_gemm_kernel<<<dim3(3 * C / TN, (B * T) / TM), 256, 0, stream>>>(
        xb, wqkvb, outK, outV, qb, kbw, vtb);

    // 2) flash attention
    attn_kernel<<<dim3(T / 64, H, B), 128, 0, stream>>>(qb, kbw, vtb, yb);

    // 3) output projection
    proj_gemm_kernel<<<dim3(C / TN, (B * T) / TM), 256, 0, stream>>>(
        yb, wprojb, outY);
}
